// SS2D_31877247270975
// MI455X (gfx1250) — matrix operand order repaired and checked by reading, not compiled, not run on hardware
//
#include <hip/hip_runtime.h>

// ---------------- constants ----------------
#define B_    32
#define N_    128
#define DM_   384
#define DI_   768
#define DS_   8
#define KDIR  4
#define HC_   16
#define WC_   24
#define L_    384          // HC_*WC_
#define EPS_  1e-5f
#define MROWS 12288        // B_*L_

typedef unsigned short ushort_t;
typedef __attribute__((ext_vector_type(16))) __bf16 v16bf;
typedef __attribute__((ext_vector_type(8)))  float  v8f;

union FragBF { uint4 q[2]; v16bf v; };

// gfx1250 async global->LDS DMA (ASYNCcnt-tracked), guarded so fallback still compiles
#if defined(__AMDGCN__) && __has_builtin(__builtin_amdgcn_global_load_async_to_lds_b128) && \
    __has_builtin(__builtin_amdgcn_s_wait_asynccnt)
#define USE_ASYNC_LDS 1
typedef __attribute__((ext_vector_type(4))) int v4i_t;
typedef __attribute__((address_space(1))) v4i_t g1_v4i;   // global
typedef __attribute__((address_space(3))) v4i_t l3_v4i;   // LDS
#else
#define USE_ASYNC_LDS 0
#endif

__device__ __forceinline__ float bf2f(ushort_t u) {
  union { unsigned u; float f; } c; c.u = ((unsigned)u) << 16; return c.f;
}
__device__ __forceinline__ ushort_t f2bf(float f) {
  union { float f; unsigned u; } c; c.f = f;
  unsigned r = c.u + 0x7FFFu + ((c.u >> 16) & 1u);
  return (ushort_t)(r >> 16);
}

// ---------------- weight conversions ----------------
__global__ void k_cvt_bf16(const float* __restrict__ src, ushort_t* __restrict__ dst, int n) {
  int i = blockIdx.x * 256 + threadIdx.x;
  if (i < n) dst[i] = f2bf(src[i]);
}

// x_proj_w [KDIR][40][768] -> padded/reordered [KDIR][64][768]:
// out col 0..23 = dt rows ; 24..31 = 0 ; 32..39 = Bm(24..31) ; 40..47 = Cm(32..39) ; 48..63 = 0
__global__ void k_cvt_xproj(const float* __restrict__ src, ushort_t* __restrict__ dst) {
  int idx = blockIdx.x * 256 + threadIdx.x;
  if (idx >= KDIR * 64 * 768) return;
  int d = idx % 768;
  int c = (idx / 768) % 64;
  int k = idx / (768 * 64);
  float v = 0.f;
  int srcc = -1;
  if (c < 24) srcc = c;
  else if (c >= 32 && c < 48) srcc = c - 8;
  if (srcc >= 0) v = src[((size_t)k * 40 + srcc) * 768 + d];
  dst[idx] = f2bf(v);
}

// dt_w rows of 24 -> zero-padded rows of 32
__global__ void k_cvt_pad(const float* __restrict__ src, ushort_t* __restrict__ dst,
                          int rows, int srcK, int dstK) {
  int idx = blockIdx.x * 256 + threadIdx.x;
  if (idx >= rows * dstK) return;
  int c = idx % dstK, r = idx / dstK;
  dst[idx] = f2bf((c < srcK) ? src[(size_t)r * srcK + c] : 0.f);
}

// ---------------- LayerNorm(384) for r/n/t -> concat bf16 [4096][1152] ----------------
__global__ void k_ln3(const float* __restrict__ r, const float* __restrict__ n,
                      const float* __restrict__ t, const float* __restrict__ g,
                      const float* __restrict__ b, ushort_t* __restrict__ lncat) {
  int row = blockIdx.x;
  int s   = blockIdx.y;
  const float* x = (s == 0) ? r : (s == 1) ? n : t;
  x += (size_t)row * DM_;
  int tid = threadIdx.x;           // 128
  __shared__ float sm[128];
  float v0 = x[tid], v1 = x[tid + 128], v2 = x[tid + 256];
  sm[tid] = v0 + v1 + v2; __syncthreads();
  for (int k = 64; k > 0; k >>= 1) { if (tid < k) sm[tid] += sm[tid + k]; __syncthreads(); }
  float mean = sm[0] / DM_; __syncthreads();
  float d0 = v0 - mean, d1 = v1 - mean, d2 = v2 - mean;
  sm[tid] = d0*d0 + d1*d1 + d2*d2; __syncthreads();
  for (int k = 64; k > 0; k >>= 1) { if (tid < k) sm[tid] += sm[tid + k]; __syncthreads(); }
  float rs = rsqrtf(sm[0] / DM_ + EPS_); __syncthreads();
  ushort_t* o = lncat + (size_t)row * 1152 + s * DM_;
  o[tid]       = f2bf(d0 * rs * g[tid]       + b[tid]);
  o[tid + 128] = f2bf(d1 * rs * g[tid + 128] + b[tid + 128]);
  o[tid + 256] = f2bf(d2 * rs * g[tid + 256] + b[tid + 256]);
}

// ---------------- WMMA bf16 GEMM:  C[m,n] = sum_k A[m,k] * W[n,k] ----------------
// Requirements: M % 128 == 0, N % 64 == 0, K % 32 == 0, 16B-aligned pointers/strides.
// Block: 128 threads / 4 waves. Block tile 128(M) x 64(N); wave tile 32 x 64 (8 WMMAs/chunk).
// B tile staged to LDS via gfx1250 async DMA (ASYNCcnt) when available.
#define BSTRIDE 40   // 32 K-elems + 8 pad (80B rows: 16B aligned, conflict-free 16-lane reads)

__global__ void __launch_bounds__(128)
k_wmma_gemm(const ushort_t* __restrict__ A, long lda,
            const ushort_t* __restrict__ W, long ldw,
            float* __restrict__ Cf, ushort_t* __restrict__ Cb,
            long ldc, int Kd) {
  __shared__ ushort_t lB[64 * BSTRIDE];
  int tid  = threadIdx.x;
  int wave = tid >> 5;
  int lane = tid & 31;
  int half = lane >> 4;
  int l16  = lane & 15;
  int mb   = blockIdx.x * 128;
  int colb = blockIdx.y * 64;

  const ushort_t* Arow0 = A + (size_t)(mb + wave * 32 + l16) * lda;
  const ushort_t* Arow1 = Arow0 + (size_t)16 * lda;
  // B stage mapping: 256 x 16B moves, 2 per thread
  int i0 = tid, i1 = tid + 128;
  int sc0 = i0 >> 2, sk0 = (i0 & 3) * 8;
  int sc1 = i1 >> 2, sk1 = (i1 & 3) * 8;
  const ushort_t* Wp0 = W + (size_t)(colb + sc0) * ldw + sk0;
  const ushort_t* Wp1 = W + (size_t)(colb + sc1) * ldw + sk1;
  ushort_t* ld0 = lB + sc0 * BSTRIDE + sk0;
  ushort_t* ld1 = lB + sc1 * BSTRIDE + sk1;

  v8f acc[2][4];
  #pragma unroll
  for (int m = 0; m < 2; m++)
    #pragma unroll
    for (int t = 0; t < 4; t++)
      #pragma unroll
      for (int i = 0; i < 8; i++) acc[m][t][i] = 0.f;

  for (int k0 = 0; k0 < Kd; k0 += 32) {
    __syncthreads();                      // protect LDS from previous iteration's readers
#if USE_ASYNC_LDS
    __builtin_amdgcn_global_load_async_to_lds_b128((g1_v4i*)(Wp0 + k0), (l3_v4i*)ld0, 0, 0);
    __builtin_amdgcn_global_load_async_to_lds_b128((g1_v4i*)(Wp1 + k0), (l3_v4i*)ld1, 0, 0);
    __builtin_amdgcn_s_wait_asynccnt(0);
#else
    *(uint4*)ld0 = *(const uint4*)(Wp0 + k0);
    *(uint4*)ld1 = *(const uint4*)(Wp1 + k0);
#endif
    __syncthreads();
    __builtin_prefetch(Arow0 + k0 + 128, 0, 3);
    // A 16x32 frag: lane-half h holds K = h*8+[0..7] (v0..3), 16+h*8+[0..7] (v4..7)
    FragBF a0, a1;
    a0.q[0] = *(const uint4*)(Arow0 + k0 + half * 8);
    a0.q[1] = *(const uint4*)(Arow0 + k0 + 16 + half * 8);
    a1.q[0] = *(const uint4*)(Arow1 + k0 + half * 8);
    a1.q[1] = *(const uint4*)(Arow1 + k0 + 16 + half * 8);
    #pragma unroll
    for (int t = 0; t < 4; t++) {
      // B 32x16 frag from LDS: lane-half h holds K = h*8+[0..7], 16+h*8+[0..7], column = lane&15
      const ushort_t* bp = lB + (t * 16 + l16) * BSTRIDE + half * 8;
      FragBF bf_;
      bf_.q[0] = *(const uint4*)(bp);
      bf_.q[1] = *(const uint4*)(bp + 16);
      acc[0][t] = __builtin_amdgcn_wmma_f32_16x16x32_bf16(
          false, a0.v, false, bf_.v, (short)0, acc[0][t], false, false);
      acc[1][t] = __builtin_amdgcn_wmma_f32_16x16x32_bf16(
          false, a1.v, false, bf_.v, (short)0, acc[1][t], false, false);
    }
  }

  // C/D layout: lanes 0-15 -> M=r, lanes 16-31 -> M=8+r ; N = lane&15
  #pragma unroll
  for (int m = 0; m < 2; m++) {
    int rb = mb + wave * 32 + m * 16 + half * 8;
    #pragma unroll
    for (int t = 0; t < 4; t++) {
      int col = colb + t * 16 + l16;
      #pragma unroll
      for (int r2 = 0; r2 < 8; r2++) {
        size_t idx = (size_t)(rb + r2) * ldc + col;
        if (Cb) Cb[idx] = f2bf(acc[m][t][r2]);
        else    Cf[idx] = acc[m][t][r2];
      }
    }
  }
}

// ---------------- depthwise 3x3 conv + BN + SiLU, concat along W ----------------
struct ConvParams { const float *w, *b, *g, *beta, *m, *v; };

__global__ void k_conv(const float* __restrict__ xz, ConvParams p0, ConvParams p1,
                       ConvParams p2, float* __restrict__ x2) {
  int bc = blockIdx.x;                 // b*768 + c
  int b = bc / DI_, c = bc % DI_;
  int tid = threadIdx.x;               // 128: h = tid>>3, w = tid&7
  int h = tid >> 3, w = tid & 7;
  __shared__ float tile[16][8];
  for (int s = 0; s < 3; s++) {
    const ConvParams& P = (s == 0) ? p0 : (s == 1) ? p1 : p2;
    tile[h][w] = xz[((size_t)(s * 4096 + b * 128) + tid) * 1536 + c];
    __syncthreads();
    float acc = P.b[c];
    #pragma unroll
    for (int ky = 0; ky < 3; ky++)
      #pragma unroll
      for (int kx = 0; kx < 3; kx++) {
        int hh = h + ky - 1, ww = w + kx - 1;
        float xv = (hh >= 0 && hh < 16 && ww >= 0 && ww < 8) ? tile[hh][ww] : 0.f;
        acc += xv * P.w[c * 9 + ky * 3 + kx];
      }
    float bn = (acc - P.m[c]) * rsqrtf(P.v[c] + EPS_) * P.g[c] + P.beta[c];
    float out = bn / (1.f + __expf(-bn));
    x2[(((size_t)b * DI_ + c) * HC_ + h) * WC_ + s * 8 + w] = out;
    __syncthreads();
  }
}

// ---------------- build 4-direction xs (bf16, [k][b][l][d]) ----------------
__global__ void k_gather_xs(const float* __restrict__ x2, ushort_t* __restrict__ xs) {
  size_t idx = (size_t)blockIdx.x * 256 + threadIdx.x;
  if (idx >= (size_t)B_ * L_ * DI_) return;
  int d = (int)(idx % DI_);
  int l = (int)((idx / DI_) % L_);
  int b = (int)(idx / ((size_t)DI_ * L_));
  const float* base = x2 + ((size_t)b * DI_ + d) * L_;
  float v0 = base[l];                                     // dir0: l = h*24+w
  float v1 = base[(l & 15) * WC_ + (l >> 4)];             // dir1: l = w*16+h
  ushort_t b0 = f2bf(v0), b1 = f2bf(v1);
  xs[(((size_t)0 * B_ + b) * L_ + l) * DI_ + d]            = b0;
  xs[(((size_t)2 * B_ + b) * L_ + (L_ - 1 - l)) * DI_ + d] = b0;
  xs[(((size_t)1 * B_ + b) * L_ + l) * DI_ + d]            = b1;
  xs[(((size_t)3 * B_ + b) * L_ + (L_ - 1 - l)) * DI_ + d] = b1;
}

// ---------------- selective scan ----------------
// x_dbl rows (stride 64): cols 0..23 dt-rank, 24..31 zero, 32..39 Bm, 40..47 Cm
__global__ void k_scan(const ushort_t* __restrict__ xs, const ushort_t* __restrict__ xdbl,
                       const ushort_t* __restrict__ dts, const float* __restrict__ A_logs,
                       const float* __restrict__ dt_b, const float* __restrict__ Ds,
                       ushort_t* __restrict__ out_y) {
  int b = blockIdx.z, k = blockIdx.y;
  int tid = threadIdx.x;                          // 256
  int d = blockIdx.x * 256 + tid;                 // gridDim.x = 3
  int kd = k * DI_ + d;
  float Arow[8];
  #pragma unroll
  for (int n = 0; n < 8; n++) Arow[n] = -__expf(A_logs[(size_t)kd * DS_ + n]);
  float dtb = dt_b[kd];
  float Dv  = Ds[kd];
  float h[8];
  #pragma unroll
  for (int n = 0; n < 8; n++) h[n] = 0.f;
  __shared__ float sBC[16][16];                   // 16 steps x (Bm[8] | Cm[8])
  const ushort_t* xd = xdbl + ((size_t)k * MROWS + (size_t)b * L_) * 64;
  const ushort_t* us = xs  + (((size_t)k * B_ + b) * L_) * DI_ + d;
  const ushort_t* dl = dts + ((size_t)k * MROWS + (size_t)b * L_) * DI_ + d;
  ushort_t*       oy = out_y + ((size_t)k * MROWS + (size_t)b * L_) * DI_ + d;
  int j0 = tid >> 4, c0 = tid & 15;
  for (int ch = 0; ch < 24; ch++) {
    __syncthreads();
    sBC[j0][c0] = bf2f(xd[(ch * 16 + j0) * 64 + 32 + c0]);
    __syncthreads();
    for (int j = 0; j < 16; j++) {
      int l = ch * 16 + j;
      float u  = bf2f(us[(size_t)l * DI_]);
      float xr = bf2f(dl[(size_t)l * DI_]) + dtb;
      float dt = (xr > 20.f) ? xr : __logf(1.f + __expf(xr));   // softplus
      float du = dt * u;
      float y = 0.f;
      #pragma unroll
      for (int n = 0; n < 8; n++) {
        h[n] = __expf(dt * Arow[n]) * h[n] + du * sBC[j][n];
        y += h[n] * sBC[j][8 + n];
      }
      oy[(size_t)l * DI_] = f2bf(y + Dv * u);
    }
  }
}

// ---------------- combine 4 directions + out-LN + SiLU(z) gate ----------------
__global__ void k_combine(const ushort_t* __restrict__ out_y, const float* __restrict__ xz,
                          const float* __restrict__ ong, const float* __restrict__ onb,
                          ushort_t* __restrict__ yact) {
  int m = blockIdx.x;                    // b*384 + l
  int b = m / L_, l = m % L_;
  int hh = l / WC_, ww = l % WC_;
  int l1 = ww * HC_ + hh;
  int tid = threadIdx.x;                 // 256
  __shared__ float sm[256];
  float vals[3];
  float sum = 0.f;
  #pragma unroll
  for (int t = 0; t < 3; t++) {
    int d = tid + t * 256;
    float v = bf2f(out_y[(((size_t)0 * B_ + b) * L_ + l) * DI_ + d])
            + bf2f(out_y[(((size_t)2 * B_ + b) * L_ + (L_ - 1 - l)) * DI_ + d])
            + bf2f(out_y[(((size_t)1 * B_ + b) * L_ + l1) * DI_ + d])
            + bf2f(out_y[(((size_t)3 * B_ + b) * L_ + (L_ - 1 - l1)) * DI_ + d]);
    vals[t] = v; sum += v;
  }
  sm[tid] = sum; __syncthreads();
  for (int s = 128; s > 0; s >>= 1) { if (tid < s) sm[tid] += sm[tid + s]; __syncthreads(); }
  float mean = sm[0] / DI_; __syncthreads();
  float vs = 0.f;
  #pragma unroll
  for (int t = 0; t < 3; t++) { float d0 = vals[t] - mean; vs += d0 * d0; }
  sm[tid] = vs; __syncthreads();
  for (int s = 128; s > 0; s >>= 1) { if (tid < s) sm[tid] += sm[tid + s]; __syncthreads(); }
  float rs = rsqrtf(sm[0] / DI_ + EPS_); __syncthreads();
  int s_ = l >> 7, n_ = l & 127;
  const float* zrow = xz + ((size_t)(s_ * 4096 + b * 128 + n_)) * 1536 + DI_;
  ushort_t* orow = yact + (size_t)m * DI_;
  #pragma unroll
  for (int t = 0; t < 3; t++) {
    int d = tid + t * 256;
    float ln = (vals[t] - mean) * rs * ong[d] + onb[d];
    float z  = zrow[d];
    orow[d] = f2bf(ln * (z / (1.f + __expf(-z))));
  }
}

// ---------------- side-LN(1152) + residual + out_proj add -> d_out ----------------
__global__ void k_final(const float* __restrict__ side_raw, const float* __restrict__ outmm,
                        const float* __restrict__ rin, const float* __restrict__ nin,
                        const float* __restrict__ tin, const float* __restrict__ slg,
                        const float* __restrict__ slb, float* __restrict__ out) {
  int row = blockIdx.x;                 // b*128 + n
  int b = row >> 7, n = row & 127;
  int tid = threadIdx.x;                // 256
  const float* srow = side_raw + (size_t)row * 1152;
  __shared__ float sm[256];
  float vloc[5];
  float sum = 0.f;
  #pragma unroll
  for (int t = 0; t < 5; t++) {
    int c = tid + t * 256;
    float v = (c < 1152) ? srow[c] : 0.f;
    vloc[t] = v; sum += v;
  }
  sm[tid] = sum; __syncthreads();
  for (int s = 128; s > 0; s >>= 1) { if (tid < s) sm[tid] += sm[tid + s]; __syncthreads(); }
  float mean = sm[0] / 1152.f; __syncthreads();
  float vs = 0.f;
  #pragma unroll
  for (int t = 0; t < 5; t++) {
    int c = tid + t * 256;
    if (c < 1152) { float d0 = vloc[t] - mean; vs += d0 * d0; }
  }
  sm[tid] = vs; __syncthreads();
  for (int s = 128; s > 0; s >>= 1) { if (tid < s) sm[tid] += sm[tid + s]; __syncthreads(); }
  float rs = rsqrtf(sm[0] / 1152.f + EPS_); __syncthreads();
  for (int t = 0; t < 5; t++) {
    int c = tid + t * 256;
    if (c >= 1152) break;
    int s_ = c / DM_, dm = c % DM_;
    float sl = (vloc[t] - mean) * rs * slg[c] + slb[c];
    const float* xin = (s_ == 0) ? rin : (s_ == 1) ? nin : tin;
    float mm = outmm[((size_t)b * L_ + s_ * N_ + n) * DM_ + dm];
    out[(size_t)s_ * (B_ * N_ * DM_) + (size_t)row * DM_ + dm] =
        xin[(size_t)row * DM_ + dm] + sl + mm;
  }
}

// ---------------- host launch ----------------
extern "C" void kernel_launch(void* const* d_in, const int* in_sizes, int n_in,
                              void* d_out, int out_size, void* d_ws, size_t ws_size,
                              hipStream_t stream) {
  (void)in_sizes; (void)n_in; (void)out_size; (void)ws_size;
  const float* r_in      = (const float*)d_in[0];
  const float* n_in_     = (const float*)d_in[1];
  const float* t_in      = (const float*)d_in[2];
  const float* norm_g    = (const float*)d_in[3];
  const float* norm_b    = (const float*)d_in[4];
  const float* in_proj_w = (const float*)d_in[5];
  const float* out_proj_w= (const float*)d_in[6];
  const float* out_norm_g= (const float*)d_in[7];
  const float* out_norm_b= (const float*)d_in[8];
  const float* side_w    = (const float*)d_in[9];
  const float* side_ln_g = (const float*)d_in[10];
  const float* side_ln_b = (const float*)d_in[11];
  const float* x_proj_w  = (const float*)d_in[12];
  const float* dt_w      = (const float*)d_in[13];
  const float* A_logs    = (const float*)d_in[14];
  const float* Ds_       = (const float*)d_in[15];
  const float* dt_b      = (const float*)d_in[16];
  ConvParams cp[3];
  for (int s = 0; s < 3; s++) {
    int o = 17 + 6 * s;
    cp[s].w    = (const float*)d_in[o + 0];
    cp[s].b    = (const float*)d_in[o + 1];
    cp[s].g    = (const float*)d_in[o + 2];
    cp[s].beta = (const float*)d_in[o + 3];
    cp[s].m    = (const float*)d_in[o + 4];
    cp[s].v    = (const float*)d_in[o + 5];
  }

  char* ws = (char*)d_ws;
  size_t off = 0;
  auto alloc = [&](size_t bytes) { size_t o = off; off += (bytes + 255) & ~(size_t)255; return o; };
  ushort_t* lncat    = (ushort_t*)(ws + alloc((size_t)4096 * 1152 * 2));
  float*    xz       = (float*)   (ws + alloc((size_t)MROWS * 1536 * 4));
  float*    x2       = (float*)   (ws + alloc((size_t)B_ * DI_ * L_ * 4));
  ushort_t* xs       = (ushort_t*)(ws + alloc((size_t)KDIR * B_ * L_ * DI_ * 2));
  ushort_t* xdbl     = (ushort_t*)(ws + alloc((size_t)KDIR * MROWS * 64 * 2));
  ushort_t* dts      = (ushort_t*)(ws + alloc((size_t)KDIR * MROWS * DI_ * 2));
  ushort_t* outy     = (ushort_t*)(ws + alloc((size_t)KDIR * MROWS * DI_ * 2));
  ushort_t* yact     = (ushort_t*)(ws + alloc((size_t)MROWS * DI_ * 2));
  float*    outmm    = (float*)   (ws + alloc((size_t)MROWS * DM_ * 4));
  float*    sider    = (float*)   (ws + alloc((size_t)4096 * 1152 * 4));
  ushort_t* w_inproj = (ushort_t*)(ws + alloc((size_t)1536 * 384 * 2));
  ushort_t* w_outproj= (ushort_t*)(ws + alloc((size_t)384 * 768 * 2));
  ushort_t* w_side   = (ushort_t*)(ws + alloc((size_t)1152 * 1152 * 2));
  ushort_t* w_xprojp = (ushort_t*)(ws + alloc((size_t)KDIR * 64 * 768 * 2));
  ushort_t* w_dtwp   = (ushort_t*)(ws + alloc((size_t)KDIR * 768 * 32 * 2));

  auto cvt = [&](const float* s, ushort_t* dptr, int nelem) {
    k_cvt_bf16<<<(nelem + 255) / 256, 256, 0, stream>>>(s, dptr, nelem);
  };
  cvt(in_proj_w,  w_inproj,  1536 * 384);
  cvt(out_proj_w, w_outproj, 384 * 768);
  cvt(side_w,     w_side,    1152 * 1152);
  k_cvt_xproj<<<(KDIR * 64 * 768 + 255) / 256, 256, 0, stream>>>(x_proj_w, w_xprojp);
  k_cvt_pad<<<(KDIR * 768 * 32 + 255) / 256, 256, 0, stream>>>(dt_w, w_dtwp,
                                                               KDIR * 768, 24, 32);

  // 1) LayerNorms -> concat bf16
  k_ln3<<<dim3(4096, 3), 128, 0, stream>>>(r_in, n_in_, t_in, norm_g, norm_b, lncat);

  // 2) in_proj (per stream, strided A): M=4096, N=1536, K=384
  for (int s = 0; s < 3; s++)
    k_wmma_gemm<<<dim3(4096 / 128, 1536 / 64), 128, 0, stream>>>(
        lncat + s * DM_, 1152, w_inproj, 384,
        xz + (size_t)s * 4096 * 1536, nullptr, 1536, 384);

  // 3) side GEMM: M=4096, N=1152, K=1152
  k_wmma_gemm<<<dim3(4096 / 128, 1152 / 64), 128, 0, stream>>>(
      lncat, 1152, w_side, 1152, sider, nullptr, 1152, 1152);

  // 4) depthwise conv branches -> x2
  k_conv<<<B_ * DI_, 128, 0, stream>>>(xz, cp[0], cp[1], cp[2], x2);

  // 5) 4-direction gather
  {
    size_t tot = (size_t)B_ * L_ * DI_;
    k_gather_xs<<<(unsigned)((tot + 255) / 256), 256, 0, stream>>>(x2, xs);
  }

  // 6) x_dbl (per direction): M=12288, N=64 (padded/reordered), K=768 -> bf16
  for (int k = 0; k < KDIR; k++)
    k_wmma_gemm<<<dim3(MROWS / 128, 1), 128, 0, stream>>>(
        xs + (size_t)k * B_ * L_ * DI_, DI_,
        w_xprojp + (size_t)k * 64 * 768, 768,
        nullptr, xdbl + (size_t)k * MROWS * 64, 64, 768);

  // 7) dts (per direction): M=12288, N=768, K=32 (zero-padded) -> bf16
  for (int k = 0; k < KDIR; k++)
    k_wmma_gemm<<<dim3(MROWS / 128, 768 / 64), 128, 0, stream>>>(
        xdbl + (size_t)k * MROWS * 64, 64,
        w_dtwp + (size_t)k * 768 * 32, 32,
        nullptr, dts + (size_t)k * MROWS * DI_, DI_, 32);

  // 8) selective scan
  k_scan<<<dim3(3, KDIR, B_), 256, 0, stream>>>(xs, xdbl, dts, A_logs, dt_b, Ds_, outy);

  // 9) combine + out-LN + SiLU gate
  k_combine<<<MROWS, 256, 0, stream>>>(outy, xz, out_norm_g, out_norm_b, yact);

  // 10) out_proj: M=12288, N=384, K=768
  k_wmma_gemm<<<dim3(MROWS / 128, 384 / 64), 128, 0, stream>>>(
      yact, DI_, w_outproj, 768, outmm, nullptr, 384, 768);

  // 11) side-LN + residuals -> d_out
  k_final<<<4096, 256, 0, stream>>>(sider, outmm, r_in, n_in_, t_in,
                                    side_ln_g, side_ln_b, (float*)d_out);
}
